// FeatureQuantizerEMA_3745211482833
// MI455X (gfx1250) — matrix operand order repaired and checked by reading, not compiled, not run on hardware
//
#include <hip/hip_runtime.h>
#include <hip/hip_bf16.h>

// ---------------------------------------------------------------- constants
#define EMB_DIM   256
#define NUM_EMB   1024
#define NROWS     131072            // B*H*W = 32*64*64
#define HW        4096              // H*W
#define OUT_ELEMS 33554432          // B*C*H*W
#define DECAY     0.9f
#define ONE_M_DECAY 0.1f
#define EPSV      1e-5f
#define COMMIT    0.25f

// d_out float offsets (outputs concatenated in return order)
#define OFF_LOSS  33554432
#define OFF_IDX   33554433
#define OFF_NEMB  33685505          // new_embed  [D,K]
#define OFF_NCS   33947649          // new_cluster_size [K]
#define OFF_NEMA  33948673          // new_ema_embed [D,K]

// ws byte offsets
#define WS_EMBT   0                 // 1024*256 ushort (bf16, code-major) = 524288 B
#define WS_ENORM  524288            // 1024 f32
#define WS_RC     528384            // 1024 f32 (ref counts)
#define WS_DW     532480            // 1024*256 f32 = 1 MB  (dw[k][d])
#define WS_IDX    1581056           // 131072 i32
#define WS_NSUM   2105344           // f32
#define WS_LOSS   2105348           // f32 (raw sum of squares)

typedef __attribute__((ext_vector_type(16))) __bf16 v16bf;
typedef __attribute__((ext_vector_type(8)))  float  v8f;

__device__ __forceinline__ unsigned short f2bf(float f) {
  unsigned int u = __float_as_uint(f);
  unsigned int r = u + 0x7FFFu + ((u >> 16) & 1u);   // round-to-nearest-even
  return (unsigned short)(r >> 16);
}

// ------------------------------------------------------------- init scratch
__global__ void vq_init(float* dw, float* rc, float* nsum, float* lossAcc) {
  int t = blockIdx.x * blockDim.x + threadIdx.x;      // 262144 threads
  dw[t] = 0.0f;
  if (t < NUM_EMB) rc[t] = 0.0f;
  if (t == 0) { *nsum = 0.0f; *lossAcc = 0.0f; }
}

// ---------------------------------------- embed -> bf16 code-major + norms²
__global__ __launch_bounds__(256)
void vq_prep_embed(const float* __restrict__ embed,
                   unsigned short* __restrict__ embT,
                   float* __restrict__ enorm) {
  __shared__ float red[256];
  const int k = blockIdx.x;          // code
  const int d = threadIdx.x;         // dim
  float v = embed[d * NUM_EMB + k];
  embT[k * EMB_DIM + d] = f2bf(v);
  red[d] = v * v;
  __syncthreads();
  for (int s = 128; s > 0; s >>= 1) {
    if (d < s) red[d] += red[d + s];
    __syncthreads();
  }
  if (d == 0) enorm[k] = red[0];
}

// --------------------------- main kernel: distance GEMM (WMMA bf16) + argmin
// block = 256 threads = 8 waves; wave handles 16 rows; block = 128 rows.
// embed tiles double-buffered in LDS via async global->LDS copies (ASYNCcnt).
#define CHUNK_CODES 64
#define NCHUNKS     16              // 1024 / 64
#define CHUNK_BYTES (CHUNK_CODES * EMB_DIM * 2)   // 32768
#define CHUNK_V4    (CHUNK_BYTES / 16)            // 2048 uint4

__device__ __forceinline__ void stage_chunk_async(const uint4* __restrict__ embT4,
                                                  unsigned sbase, int c, int buf,
                                                  int tid) {
  const uint4* src = embT4 + c * CHUNK_V4 + tid;
  const unsigned dst = sbase + (unsigned)(buf * CHUNK_BYTES + tid * 16);
#pragma unroll
  for (int j = 0; j < 8; ++j) {
    asm volatile("global_load_async_to_lds_b128 %0, %1, off"
                 :: "v"(dst + j * 4096), "v"(src + j * 256)
                 : "memory");
  }
}

__global__ __launch_bounds__(256)
void vq_dist_argmin(const float* __restrict__ x,
                    const unsigned short* __restrict__ embT,
                    const float* __restrict__ enorm,
                    float* __restrict__ idxF,
                    int* __restrict__ idxI,
                    float* __restrict__ refcount) {
  __shared__ unsigned short sB[2 * CHUNK_CODES * EMB_DIM];   // 64 KB, 2 buffers

  const int tid  = threadIdx.x;
  const int lane = tid & 31;
  const int wave = tid >> 5;
  const int half = lane >> 4;                    // 0: lanes 0-15, 1: lanes 16-31
  const int l16  = lane & 15;
  const int rowbase = blockIdx.x * 128 + wave * 16;
  const int row  = rowbase + l16;                // A row for this lane (M = lane%16)
  const long xbase = (long)(row >> 12) * (EMB_DIM * HW) + (row & (HW - 1));
  const uint4* embT4 = reinterpret_cast<const uint4*>(embT);
  const unsigned sbase = (unsigned)(unsigned long long)(const void*)sB;  // LDS byte offset

  // kick off first chunk's async staging before the (long) A-load phase
  stage_chunk_async(embT4, sbase, 0, 0, tid);

  // ---- load A = (-2*x) in bf16, resident for whole kernel (contiguous k = 16*half + e, same as B)
  v16bf a[8];
#pragma unroll
  for (int ks = 0; ks < 8; ++ks) {
    union { v16bf v; unsigned short u[16]; } av;
    const int d0 = ks * 32 + (half ? 16 : 0);
#pragma unroll
    for (int e = 0; e < 16; ++e) {
      const int d = d0 + e;
      av.u[e] = f2bf(-2.0f * x[xbase + (long)d * HW]);   // coalesced per half-wave
    }
    a[ks] = av.v;
  }

  float minv[8];
  int   mini[8];
#pragma unroll
  for (int r = 0; r < 8; ++r) { minv[r] = 3.0e38f; mini[r] = 0; }

  for (int chunk = 0; chunk < NCHUNKS; ++chunk) {
    const int buf = chunk & 1;
    if (chunk + 1 < NCHUNKS) {
      // prefetch next chunk into the other buffer (its last reader already
      // passed the barrier at the end of iteration chunk-1)
      stage_chunk_async(embT4, sbase, chunk + 1, buf ^ 1, tid);
      asm volatile("s_wait_asynccnt 0x8" ::: "memory");  // chunk's 8 retired (in-order)
    } else {
      asm volatile("s_wait_asynccnt 0x0" ::: "memory");
    }
    __syncthreads();                 // all waves' async writes for `chunk` visible

#pragma unroll
    for (int nt = 0; nt < 4; ++nt) {
      const int nglob = chunk * CHUNK_CODES + nt * 16 + l16;  // code (N) of this lane
      const float en = enorm[nglob];
      v8f acc;
#pragma unroll
      for (int r = 0; r < 8; ++r) acc[r] = en;           // seed with ||e||^2
      const unsigned short* bp =
          sB + buf * (CHUNK_CODES * EMB_DIM) + (nt * 16 + l16) * EMB_DIM +
          (half ? 16 : 0);
#pragma unroll
      for (int ks = 0; ks < 8; ++ks) {
        v16bf b = *reinterpret_cast<const v16bf*>(bp + ks * 32);  // 32B contiguous LDS
        acc = __builtin_amdgcn_wmma_f32_16x16x32_bf16(
            false, a[ks], false, b, (short)0, acc, false, false);
      }
#pragma unroll
      for (int r = 0; r < 8; ++r) {
        if (acc[r] < minv[r]) { minv[r] = acc[r]; mini[r] = nglob; }
      }
    }
    __syncthreads();                 // done reading `buf` before it is re-staged
  }

  // ---- argmin across the 16 lanes sharing an M row (halves independent)
#pragma unroll
  for (int r = 0; r < 8; ++r) {
    float mv = minv[r]; int mi = mini[r];
#pragma unroll
    for (int m = 1; m <= 8; m <<= 1) {
      float ov = __shfl_xor(mv, m, 32);
      int   oi = __shfl_xor(mi, m, 32);
      if (ov < mv || (ov == mv && oi < mi)) { mv = ov; mi = oi; }
    }
    if (l16 == 0) {                                  // lane 0 -> M=r, lane 16 -> M=8+r
      const int orow = rowbase + r + (half ? 8 : 0);
      idxF[orow] = (float)mi;
      idxI[orow] = mi;
      atomicAdd(&refcount[mi], 1.0f);
    }
  }
}

// ------------------------ gather quantize, write NCHW out, accumulate loss
__global__ __launch_bounds__(256)
void vq_quant_out_loss(const float* __restrict__ x,
                       const float* __restrict__ embed,
                       const int* __restrict__ idxI,
                       float* __restrict__ out,
                       float* __restrict__ lossAcc) {
  __shared__ float red[256];
  float s = 0.0f;
  for (int o = blockIdx.x * 256 + threadIdx.x; o < OUT_ELEMS;
       o += gridDim.x * 256) {
    const int p = o & (HW - 1);
    const int c = (o >> 12) & 255;
    const int b = o >> 20;
    const int i = (b << 12) | p;
    const int k = idxI[i];
    const float q  = embed[c * NUM_EMB + k];
    const float xv = x[o];
    const float d  = q - xv;
    out[o] = q;                       // straight-through: out == quantize
    s += d * d;
  }
  red[threadIdx.x] = s;
  __syncthreads();
  for (int st = 128; st > 0; st >>= 1) {
    if (threadIdx.x < st) red[threadIdx.x] += red[threadIdx.x + st];
    __syncthreads();
  }
  if (threadIdx.x == 0) atomicAdd(lossAcc, red[0]);
}

// ------------------------------------------- dw[k][d] = sum of rows per code
__global__ __launch_bounds__(256)
void vq_dw_scatter(const float* __restrict__ x,
                   const int* __restrict__ idxI,
                   float* __restrict__ dw) {
  const int i = blockIdx.x * 256 + threadIdx.x;   // row
  const int d = blockIdx.y;                       // channel
  const int b = i >> 12;
  const float xv = x[((long)b << 20) + ((long)d << 12) + (i & (HW - 1))];
  const int k = idxI[i];
  atomicAdd(&dw[k * EMB_DIM + d], xv);
}

// ------------- new_cluster_size, n = sum(ncs), final loss scale
__global__ __launch_bounds__(1024)
void vq_ncs(const float* __restrict__ cs,
            const float* __restrict__ rc,
            float* __restrict__ out,
            float* __restrict__ nsum,
            const float* __restrict__ lossAcc) {
  __shared__ float red[1024];
  const int k = threadIdx.x;
  const float ncs = cs[k] * DECAY + ONE_M_DECAY * rc[k];
  out[OFF_NCS + k] = ncs;
  red[k] = ncs;
  __syncthreads();
  for (int s = 512; s > 0; s >>= 1) {
    if (k < s) red[k] += red[k + s];
    __syncthreads();
  }
  if (k == 0) {
    *nsum = red[0];
    out[OFF_LOSS] = COMMIT * (*lossAcc) / (float)OUT_ELEMS;
  }
}

// -------------------- new_ema_embed + new_embed (elementwise, [D,K] major)
__global__ __launch_bounds__(256)
void vq_final(const float* __restrict__ ema,
              const float* __restrict__ dw,
              float* __restrict__ out,
              const float* __restrict__ nsum) {
  const int t = blockIdx.x * 256 + threadIdx.x;   // 262144 = D*K, t = d*K + k
  const int k = t & (NUM_EMB - 1);
  const int d = t >> 10;
  const float n = *nsum;
  const float newema = ema[t] * DECAY + ONE_M_DECAY * dw[k * EMB_DIM + d];
  const float ncs = out[OFF_NCS + k];
  const float smoothing = n * ((ncs + EPSV) / (n + ncs * EPSV));
  out[OFF_NEMA + t] = newema;
  out[OFF_NEMB + t] = newema / smoothing;
}

// --------------------------------------------------------------------------
extern "C" void kernel_launch(void* const* d_in, const int* in_sizes, int n_in,
                              void* d_out, int out_size, void* d_ws, size_t ws_size,
                              hipStream_t stream) {
  const float* x     = (const float*)d_in[0];
  const float* embed = (const float*)d_in[1];
  const float* cs    = (const float*)d_in[2];
  const float* ema   = (const float*)d_in[3];
  float* out = (float*)d_out;
  char*  ws  = (char*)d_ws;

  unsigned short* embT = (unsigned short*)(ws + WS_EMBT);
  float* enorm   = (float*)(ws + WS_ENORM);
  float* rc      = (float*)(ws + WS_RC);
  float* dw      = (float*)(ws + WS_DW);
  int*   idxI    = (int*)(ws + WS_IDX);
  float* nsum    = (float*)(ws + WS_NSUM);
  float* lossAcc = (float*)(ws + WS_LOSS);

  vq_init<<<1024, 256, 0, stream>>>(dw, rc, nsum, lossAcc);
  vq_prep_embed<<<NUM_EMB, 256, 0, stream>>>(embed, embT, enorm);
  vq_dist_argmin<<<NROWS / 128, 256, 0, stream>>>(x, embT, enorm,
                                                  out + OFF_IDX, idxI, rc);
  vq_quant_out_loss<<<4096, 256, 0, stream>>>(x, embed, idxI, out, lossAcc);
  dim3 gdw(NROWS / 256, EMB_DIM);
  vq_dw_scatter<<<gdw, 256, 0, stream>>>(x, idxI, dw);
  vq_ncs<<<1, 1024, 0, stream>>>(cs, rc, out, nsum, lossAcc);
  vq_final<<<(EMB_DIM * NUM_EMB) / 256, 256, 0, stream>>>(ema, dw, out, nsum);
}
